// SelfAttention_16011638080056
// MI455X (gfx1250) — matrix operand order repaired and checked by reading, not compiled, not run on hardware
//
#include <hip/hip_runtime.h>

typedef __attribute__((ext_vector_type(16))) _Float16 v16h;
typedef __attribute__((ext_vector_type(8)))  _Float16 v8h;
typedef __attribute__((ext_vector_type(8)))  float    v8f;
typedef _Float16 half_t;

#define BATCH  4
#define NPOS   4096
#define CDIM   256
#define C8     32
#define KBLK   64            // keys/values per attention block
#define NBLK   (NPOS / KBLK) // 64 blocks
#define QCHUNK 128           // query rows per workgroup (8 waves x 16)
#define NOTILE 20            // projection output tiles: f:2 g:2 h:16
#define LOG2E  1.4426950408889634f

// ---------------------------------------------------------------------------
// Kernel 0: pack kf/kg/kh into the per-lane WMMA B-matrix layout (f16) so the
// projection kernel loads one 32B vector per K-step.
// ---------------------------------------------------------------------------
__global__ __launch_bounds__(256)
void pack_w_kernel(const float* __restrict__ kf,
                   const float* __restrict__ kg,
                   const float* __restrict__ kh,
                   half_t* __restrict__ Wp)
{
    const int idx = blockIdx.x * 256 + threadIdx.x;   // (ot, ks, lane)
    if (idx >= NOTILE * 8 * 32) return;
    const int ot   = idx >> 8;
    const int ks   = (idx >> 5) & 7;
    const int lane = idx & 31;
    const int l16  = lane & 15;
    const int kb   = (lane >> 4) * 16;

    const float* W; int outW, col0;
    if (ot < 2)      { W = kf; outW = C8;   col0 = ot * 16; }
    else if (ot < 4) { W = kg; outW = C8;   col0 = (ot - 2) * 16; }
    else             { W = kh; outW = CDIM; col0 = (ot - 4) * 16; }

    half_t* dst = Wp + (size_t)idx * 16;
    #pragma unroll
    for (int i = 0; i < 16; ++i)
        dst[i] = (_Float16)W[(size_t)(ks * 32 + kb + i) * outW + col0 + l16];
}

// ---------------------------------------------------------------------------
// Kernel 1: fused 1x1-conv projections  f = x@Kf, g = x@Kg, h = x@Kh
//   f,g -> f16 row-major [B*N][32]
//   h   -> f16 blocked-transposed [B][NBLK][C][64]  (value blocks, coalesced)
// ---------------------------------------------------------------------------
__global__ __launch_bounds__(256)
void proj_kernel(const float* __restrict__ x,
                 const half_t* __restrict__ Wp,
                 half_t* __restrict__ fK,
                 half_t* __restrict__ gQ,
                 half_t* __restrict__ hB)
{
    const int tid  = threadIdx.x;
    const int lane = tid & 31;
    const int wave = tid >> 5;
    const int l16  = lane & 15;
    const int hi   = lane >> 4;

    const int row0 = blockIdx.x * 16;    // global row in [0, B*N)
    const int b    = row0 / NPOS;
    const int pos0 = row0 % NPOS;

    // 16x256 activation tile -> A-layout f16, 8 K-steps
    v16h A[8];
    const float* xrow = x + (size_t)(row0 + l16) * CDIM + hi * 16;
    #pragma unroll
    for (int ks = 0; ks < 8; ++ks) {
        const float* p = xrow + ks * 32;
        v16h a;
        #pragma unroll
        for (int i = 0; i < 8; ++i) {
            a[i]     = (_Float16)p[i];
            a[i + 8] = (_Float16)p[i + 8];
        }
        A[ks] = a;
    }

    for (int ot = wave; ot < NOTILE; ot += 8) {
        const v16h* wtile = (const v16h*)(Wp + ((size_t)ot * 8 * 32 + lane) * 16);
        v16h bm[8];
        #pragma unroll
        for (int ks = 0; ks < 8; ++ks) bm[ks] = wtile[(size_t)ks * 32];

        v8f acc = {0.f, 0.f, 0.f, 0.f, 0.f, 0.f, 0.f, 0.f};
        #pragma unroll
        for (int ks = 0; ks < 8; ++ks)
            acc = __builtin_amdgcn_wmma_f32_16x16x32_f16(false, A[ks], false, bm[ks],
                                                         (short)0, acc, false, false);

        if (ot >= 4) {
            const int col0 = (ot - 4) * 16;
            const int jb   = pos0 >> 6;
            const int pin0 = (pos0 & 63) + hi * 8;
            v8h d;
            #pragma unroll
            for (int v = 0; v < 8; ++v) d[v] = (_Float16)acc[v];
            half_t* dst = hB + (((size_t)b * NBLK + jb) * CDIM + (col0 + l16)) * KBLK + pin0;
            *(v8h*)dst = d;
        } else {
            const int col0 = (ot & 1) * 16;
            half_t* dst = (ot < 2 ? fK : gQ)
                        + (size_t)(row0 + hi * 8) * C8 + (col0 + l16);
            #pragma unroll
            for (int v = 0; v < 8; ++v) dst[(size_t)v * C8] = (_Float16)acc[v];
        }
    }
}

// ---------------------------------------------------------------------------
// Kernel 2: flash attention  out = gamma * softmax(g fᵀ) h + x
// 8 waves, 128 query rows/WG, 64-key blocks, base-2 online softmax,
// async global->LDS double-buffered staging.
// ---------------------------------------------------------------------------
__global__ __launch_bounds__(256)
void attn_kernel(const half_t* __restrict__ gQ,
                 const half_t* __restrict__ fK,
                 const half_t* __restrict__ hB,
                 const float*  __restrict__ x,
                 const float*  __restrict__ gammap,
                 float* __restrict__ out)
{
    __shared__ __align__(32) half_t fS[2][KBLK * C8];     // key blocks   2 x 4 KB
    __shared__ __align__(32) half_t hS[2][CDIM * KBLK];   // value blocks 2 x 32 KB
    __shared__ __align__(32) half_t pS[8][16 * KBLK];     // p bounce     16 KB

    const int tid  = threadIdx.x;
    const int lane = tid & 31;
    const int wave = tid >> 5;
    const int l16  = lane & 15;
    const int hi   = lane >> 4;

    const int b  = blockIdx.y;
    const int q0 = blockIdx.x * QCHUNK + wave * 16;

    // Async global->LDS staging (LDS addr = low 32 bits of generic pointer)
    auto issue_stage = [&](int jb, int buf) {
        unsigned long long fsrc = (unsigned long long)(uintptr_t)
            ((const char*)(fK + (size_t)(b * NPOS + jb * KBLK) * C8) + (size_t)tid * 16);
        unsigned flds = (unsigned)(uintptr_t)((const char*)&fS[buf][0] + tid * 16);
        asm volatile("global_load_async_to_lds_b128 %0, %1, off"
                     :: "v"(flds), "v"(fsrc) : "memory");
        unsigned long long hsrc = (unsigned long long)(uintptr_t)
            ((const char*)(hB + ((size_t)b * NBLK + jb) * CDIM * KBLK) + (size_t)tid * 128);
        unsigned hlds = (unsigned)(uintptr_t)((const char*)&hS[buf][0] + tid * 128);
        #pragma unroll
        for (int i = 0; i < 8; ++i) {
            asm volatile("global_load_async_to_lds_b128 %0, %1, off"
                         :: "v"(hlds + 16u * i), "v"(hsrc + 16ull * i) : "memory");
        }
    };

    // Query A-tile, pre-scaled by log2(e) so softmax uses native exp2
    v16h qA;
    {
        const half_t* gp = gQ + (size_t)(b * NPOS + q0 + l16) * C8;
        v8h lo  = *(const v8h*)(gp + hi * 16);
        v8h hiv = *(const v8h*)(gp + 8 + hi * 16);
        const _Float16 l2e = (_Float16)LOG2E;
        #pragma unroll
        for (int i = 0; i < 8; ++i) { qA[i] = lo[i] * l2e; qA[i + 8] = hiv[i] * l2e; }
    }

    v8f acc[16];
    #pragma unroll
    for (int t = 0; t < 16; ++t)
        acc[t] = (v8f){0.f, 0.f, 0.f, 0.f, 0.f, 0.f, 0.f, 0.f};
    float mrun[8], lrun[8];
    #pragma unroll
    for (int v = 0; v < 8; ++v) { mrun[v] = -1e30f; lrun[v] = 0.f; }

    half_t* pw = &pS[wave][0];

    issue_stage(0, 0);

    for (int jb = 0; jb < NBLK; ++jb) {
        const int buf = jb & 1;
        asm volatile("s_wait_asynccnt 0" ::: "memory");
        __syncthreads();
        if (jb + 1 < NBLK) issue_stage(jb + 1, buf ^ 1);

        const half_t* fb = &fS[buf][0];
        const half_t* hb = &hS[buf][0];

        // s tile: 16 queries x 64 keys (4 WMMAs, K = C8 = 32)
        v8f z = {0.f, 0.f, 0.f, 0.f, 0.f, 0.f, 0.f, 0.f};
        v8f s[4];
        #pragma unroll
        for (int t = 0; t < 4; ++t) {
            v16h bf = *(const v16h*)&fb[(t * 16 + l16) * C8 + hi * 16];
            s[t] = __builtin_amdgcn_wmma_f32_16x16x32_f16(false, qA, false, bf,
                                                          (short)0, z, false, false);
        }

        // Base-2 online softmax; row (hi*8+v) spans the 16 lanes of one half.
        float scl[8];
        #pragma unroll
        for (int v = 0; v < 8; ++v) {
            float rm = fmaxf(fmaxf(s[0][v], s[1][v]), fmaxf(s[2][v], s[3][v]));
            rm = fmaxf(rm, __shfl_xor(rm, 1, 32));
            rm = fmaxf(rm, __shfl_xor(rm, 2, 32));
            rm = fmaxf(rm, __shfl_xor(rm, 4, 32));
            rm = fmaxf(rm, __shfl_xor(rm, 8, 32));
            float mn = fmaxf(mrun[v], rm);
            float sc = exp2f(mrun[v] - mn);          // == 1.0 when max unchanged
            float p0 = exp2f(s[0][v] - mn);
            float p1 = exp2f(s[1][v] - mn);
            float p2 = exp2f(s[2][v] - mn);
            float p3 = exp2f(s[3][v] - mn);
            float rs = (p0 + p1) + (p2 + p3);
            rs += __shfl_xor(rs, 1, 32);
            rs += __shfl_xor(rs, 2, 32);
            rs += __shfl_xor(rs, 4, 32);
            rs += __shfl_xor(rs, 8, 32);
            lrun[v] = lrun[v] * sc + rs;
            mrun[v] = mn;
            scl[v]  = sc;
            s[0][v] = p0; s[1][v] = p1; s[2][v] = p2; s[3][v] = p3;
        }
        // Rescale the 16x256 accumulator only when some row max moved.
        int need = 0;
        #pragma unroll
        for (int v = 0; v < 8; ++v) need |= (scl[v] != 1.0f);
        if (__any(need)) {
            #pragma unroll
            for (int t = 0; t < 16; ++t)
                #pragma unroll
                for (int v = 0; v < 8; ++v) acc[t][v] *= scl[v];
        }

        // Transpose p (C/D layout -> A layout) through wave-private LDS.
        #pragma unroll
        for (int v = 0; v < 8; ++v) {
            const int r = (hi * 8 + v) * KBLK;
            pw[r + l16]      = (_Float16)s[0][v];
            pw[r + 16 + l16] = (_Float16)s[1][v];
            pw[r + 32 + l16] = (_Float16)s[2][v];
            pw[r + 48 + l16] = (_Float16)s[3][v];
        }
        asm volatile("s_wait_dscnt 0" ::: "memory");
        v16h pA0, pA1;
        {
            const half_t* pr = pw + l16 * KBLK + hi * 16;
            v8h a0 = *(const v8h*)(pr);
            v8h a1 = *(const v8h*)(pr + 8);
            v8h a2 = *(const v8h*)(pr + 32);
            v8h a3 = *(const v8h*)(pr + 40);
            #pragma unroll
            for (int i = 0; i < 8; ++i) {
                pA0[i] = a0[i]; pA0[i + 8] = a1[i];
                pA1[i] = a2[i]; pA1[i + 8] = a3[i];
            }
        }

        // o += p @ h_block : 16 channel tiles x 2 K-steps of 32
        #pragma unroll
        for (int t = 0; t < 16; ++t) {
            v16h bh0 = *(const v16h*)&hb[(t * 16 + l16) * KBLK + hi * 16];
            v16h bh1 = *(const v16h*)&hb[(t * 16 + l16) * KBLK + 32 + hi * 16];
            acc[t] = __builtin_amdgcn_wmma_f32_16x16x32_f16(false, pA0, false, bh0,
                                                            (short)0, acc[t], false, false);
            acc[t] = __builtin_amdgcn_wmma_f32_16x16x32_f16(false, pA1, false, bh1,
                                                            (short)0, acc[t], false, false);
        }
    }

    // Epilogue: out = gamma * (acc / l) + x
    const float gm = gammap[0];
    float inv[8];
    #pragma unroll
    for (int v = 0; v < 8; ++v) inv[v] = 1.0f / lrun[v];
    #pragma unroll
    for (int t = 0; t < 16; ++t) {
        const int ch = t * 16 + l16;
        #pragma unroll
        for (int v = 0; v < 8; ++v) {
            const size_t idx = ((size_t)b * NPOS + (q0 + hi * 8 + v)) * CDIM + ch;
            out[idx] = gm * (acc[t][v] * inv[v]) + x[idx];
        }
    }
}

// ---------------------------------------------------------------------------
extern "C" void kernel_launch(void* const* d_in, const int* in_sizes, int n_in,
                              void* d_out, int out_size, void* d_ws, size_t ws_size,
                              hipStream_t stream) {
    (void)in_sizes; (void)n_in; (void)out_size; (void)ws_size;
    const float* x     = (const float*)d_in[0];
    const float* kf    = (const float*)d_in[1];
    const float* kg    = (const float*)d_in[2];
    const float* kh    = (const float*)d_in[3];
    const float* gamma = (const float*)d_in[4];
    float* out = (float*)d_out;

    char* ws = (char*)d_ws;
    half_t* fK = (half_t*)(ws);                       // 1 MB   [B*N][32]
    half_t* gQ = (half_t*)(ws + (1u << 20));          // 1 MB   [B*N][32]
    half_t* hB = (half_t*)(ws + (2u << 20));          // 8 MB   [B][64][256][64]
    half_t* Wp = (half_t*)(ws + (10u << 20));         // 160 KB packed weights

    pack_w_kernel<<<dim3((NOTILE * 8 * 32 + 255) / 256), dim3(256), 0, stream>>>(
        kf, kg, kh, Wp);
    proj_kernel<<<dim3((BATCH * NPOS) / 16), dim3(256), 0, stream>>>(
        x, Wp, fK, gQ, hB);
    attn_kernel<<<dim3(NPOS / QCHUNK, BATCH), dim3(256), 0, stream>>>(
        gQ, fK, hB, x, gamma, out);
}
